// OpenGraph_69561290326842
// MI455X (gfx1250) — matrix operand order repaired and checked by reading, not compiled, not run on hardware
//
#include <hip/hip_runtime.h>
#include <hip/hip_bf16.h>
#include <stdint.h>

#define EDIM 512
#define HEADS 8
#define HD 64
#define ANCH 256
#define LGT 2
#define LN_EPS 1e-5f

typedef __bf16 bf16;
typedef bf16  v16bf __attribute__((ext_vector_type(16)));
typedef float v8f   __attribute__((ext_vector_type(8)));
typedef int   v4i   __attribute__((ext_vector_type(4)));

union FragBF { v16bf v; uint4 q[2]; };

// ---- CDNA5 async global->LDS path (ASYNCcnt), with synchronous fallback ----
#if defined(__has_builtin)
#if __has_builtin(__builtin_amdgcn_global_load_async_to_lds_b128)
#define HAVE_ASYNC_LDS 1
#endif
#endif
#ifndef HAVE_ASYNC_LDS
#define HAVE_ASYNC_LDS 0
#endif

#define ASG __attribute__((address_space(1)))
#define ASL __attribute__((address_space(3)))

__device__ __forceinline__ void async_ld16(const bf16* g, bf16* l) {
#if HAVE_ASYNC_LDS
    __builtin_amdgcn_global_load_async_to_lds_b128(
        (ASG v4i*)(ASG void*)g, (ASL v4i*)(ASL void*)l, 0, 0);
#else
    *(uint4*)l = *(const uint4*)g;
#endif
}
__device__ __forceinline__ void wait_async4() {
#if HAVE_ASYNC_LDS
#if __has_builtin(__builtin_amdgcn_s_wait_asynccnt)
    __builtin_amdgcn_s_wait_asynccnt(4);
#else
    asm volatile("s_wait_asynccnt 0x4" ::: "memory");
#endif
#endif
}
__device__ __forceinline__ void wait_async0() {
#if HAVE_ASYNC_LDS
#if __has_builtin(__builtin_amdgcn_s_wait_asynccnt)
    __builtin_amdgcn_s_wait_asynccnt(0);
#else
    asm volatile("s_wait_asynccnt 0x0" ::: "memory");
#endif
#endif
}

// WMMA issue with immediate reuse-A hint (builtin requires constant args)
#define WMMA_ACC(ACC, AV, BV, RA)                                              \
    ACC = __builtin_amdgcn_wmma_f32_16x16x32_bf16(false, AV, false, BV,        \
                                                  (short)0, ACC, RA, false)

// ---------------- f32 -> bf16 conversion (vectorized) ----------------
__global__ void conv_bf16_kernel(const float* __restrict__ in, bf16* __restrict__ out, int n) {
    int i = (blockIdx.x * 256 + threadIdx.x) * 4;
    if (i < n) {
        float4 v = *(const float4*)(in + i);
        union { bf16 h[4]; ushort4 u; } p;
        p.h[0] = (bf16)v.x; p.h[1] = (bf16)v.y;
        p.h[2] = (bf16)v.z; p.h[3] = (bf16)v.w;
        *(ushort4*)(out + i) = p.u;
    }
}

// ---------------- LayerNorm (optional residual + affine) ----------------
__global__ __launch_bounds__(256) void ln_kernel(
    const float* __restrict__ a, const float* __restrict__ res,
    const float* __restrict__ g, const float* __restrict__ b,
    float* __restrict__ out, float invscale) {
    __shared__ float s1[256], s2[256];
    int row = blockIdx.x, t = threadIdx.x;
    const float* pa = a + (size_t)row * EDIM;
    float v0 = pa[t], v1 = pa[t + 256];
    if (res) {
        const float* pr = res + (size_t)row * EDIM;
        v0 += pr[t]; v1 += pr[t + 256];
    }
    s1[t] = v0 + v1;
    s2[t] = v0 * v0 + v1 * v1;
    __syncthreads();
    for (int s = 128; s > 0; s >>= 1) {
        if (t < s) { s1[t] += s1[t + s]; s2[t] += s2[t + s]; }
        __syncthreads();
    }
    float mu  = s1[0] * (1.0f / EDIM);
    float var = s2[0] * (1.0f / EDIM) - mu * mu;
    float rs  = rsqrtf(var + LN_EPS);
    float y0 = (v0 - mu) * rs, y1 = (v1 - mu) * rs;
    if (g) { y0 = y0 * g[t] + b[t]; y1 = y1 * g[t + 256] + b[t + 256]; }
    float* po = out + (size_t)row * EDIM;
    po[t] = y0 * invscale;
    po[t + 256] = y1 * invscale;
}

// ---------------- edge-parallel SpMM via fp32 global atomics ----------------
__global__ void spmm_kernel(const float* __restrict__ x, const float* __restrict__ vals,
                            const int* __restrict__ rows, const int* __restrict__ cols,
                            float* __restrict__ out, int nnz) {
    long long tid = (long long)blockIdx.x * 256 + threadIdx.x;
    int e  = (int)(tid >> 6);
    int c8 = (int)(tid & 63);
    if (e >= nnz) return;
    float v = vals[e];
    const float* src = x + (size_t)cols[e] * EDIM + c8 * 8;
    float* dst = out + (size_t)rows[e] * EDIM + c8 * 8;
    float4 s0 = *(const float4*)src;
    float4 s1 = *(const float4*)(src + 4);
    __hip_atomic_fetch_add(dst + 0, v * s0.x, __ATOMIC_RELAXED, __HIP_MEMORY_SCOPE_AGENT);
    __hip_atomic_fetch_add(dst + 1, v * s0.y, __ATOMIC_RELAXED, __HIP_MEMORY_SCOPE_AGENT);
    __hip_atomic_fetch_add(dst + 2, v * s0.z, __ATOMIC_RELAXED, __HIP_MEMORY_SCOPE_AGENT);
    __hip_atomic_fetch_add(dst + 3, v * s0.w, __ATOMIC_RELAXED, __HIP_MEMORY_SCOPE_AGENT);
    __hip_atomic_fetch_add(dst + 4, v * s1.x, __ATOMIC_RELAXED, __HIP_MEMORY_SCOPE_AGENT);
    __hip_atomic_fetch_add(dst + 5, v * s1.y, __ATOMIC_RELAXED, __HIP_MEMORY_SCOPE_AGENT);
    __hip_atomic_fetch_add(dst + 6, v * s1.z, __ATOMIC_RELAXED, __HIP_MEMORY_SCOPE_AGENT);
    __hip_atomic_fetch_add(dst + 7, v * s1.w, __ATOMIC_RELAXED, __HIP_MEMORY_SCOPE_AGENT);
}

// ---------------- elementwise add ----------------
__global__ void add2_kernel(float* __restrict__ out, const float* __restrict__ a,
                            const float* __restrict__ b, int n) {
    int i = blockIdx.x * 256 + threadIdx.x;
    if (i < n) out[i] = a[i] + b[i];
}

// ---------------- anchor gather ----------------
__global__ void gather_kernel(const float* __restrict__ x, const int* __restrict__ idx,
                              float* __restrict__ out) {
    int i = blockIdx.x * 256 + threadIdx.x;
    int a = i >> 9, c = i & 511;
    out[i] = x[(size_t)idx[a] * EDIM + c];
}

// ---------------- bf16 WMMA GEMM: C[M,512] = A[M,512] * W[512,512]^T ----------------
// Block tile 128(M) x 128(N); 8 waves as 4x2; wave tile 32x64 -> 8 WMMA / K-step.
// Double-buffered LDS staged with GLOBAL_LOAD_ASYNC_TO_LDS_B128 (ASYNCcnt pipeline).
// A rows must be padded/zeroed to a 128 multiple by the caller (no OOB guard on loads).
__global__ __launch_bounds__(256) void gemm_bf16_kernel(
    const bf16* __restrict__ A, const bf16* __restrict__ W,
    float* __restrict__ C, int M, const float* __restrict__ bias, int relu) {
    __shared__ bf16 Asub[2][128][32];  // 16KB
    __shared__ bf16 Bsub[2][128][32];  // 16KB
    int t = threadIdx.x;
    int wave = t >> 5, lane = t & 31;
    int wr = wave >> 1, wc = wave & 1;
    int mBase = blockIdx.x * 128;
    int oBase = blockIdx.y * 128;
    // staging map: 2 threads/row, 16 bf16 (2 x b128) each, for both A and B tiles
    int ar = t >> 1, ac = (t & 1) * 16;
    const bf16* gA = A + (size_t)(mBase + ar) * EDIM + ac;
    const bf16* gW = W + (size_t)(oBase + ar) * EDIM + ac;

    v8f acc[2][4] = {};
    int halfSel = (lane < 16) ? 0 : 1;
    int mrow0 = wr * 32 + (lane & 15);

    // prologue: stage K-tile 0 into buffer 0
    async_ld16(gA,     &Asub[0][ar][ac]);
    async_ld16(gA + 8, &Asub[0][ar][ac + 8]);
    async_ld16(gW,     &Bsub[0][ar][ac]);
    async_ld16(gW + 8, &Bsub[0][ar][ac + 8]);

    for (int kk = 0; kk < EDIM; kk += 32) {
        int cur = (kk >> 5) & 1;
        if (kk + 32 < EDIM) {
            int nxt = cur ^ 1;
            async_ld16(gA + kk + 32, &Asub[nxt][ar][ac]);
            async_ld16(gA + kk + 40, &Asub[nxt][ar][ac + 8]);
            async_ld16(gW + kk + 32, &Bsub[nxt][ar][ac]);
            async_ld16(gW + kk + 40, &Bsub[nxt][ar][ac + 8]);
            wait_async4();   // retire tile cur's loads; this iteration's 4 stay in flight
        } else {
            wait_async0();
        }
        __syncthreads();

        // A fragments (two 16-row tiles): lanes 0-15 K[0..8)+[16..24), lanes 16-31 K[8..16)+[24..32)
        FragBF af0, af1;
        af0.q[0] = *(const uint4*)&Asub[cur][mrow0][halfSel * 8];
        af0.q[1] = *(const uint4*)&Asub[cur][mrow0][16 + halfSel * 8];
        af1.q[0] = *(const uint4*)&Asub[cur][mrow0 + 16][halfSel * 8];
        af1.q[1] = *(const uint4*)&Asub[cur][mrow0 + 16][16 + halfSel * 8];
        // B fragments: lane holds one column; lanes 0-15 K[0..8)+[16..24), lanes 16-31 K[8..16)+[24..32)
        FragBF bfr[4];
#pragma unroll
        for (int c = 0; c < 4; ++c) {
            int bcol = wc * 64 + c * 16 + (lane & 15);
            bfr[c].q[0] = *(const uint4*)&Bsub[cur][bcol][halfSel * 8];
            bfr[c].q[1] = *(const uint4*)&Bsub[cur][bcol][16 + halfSel * 8];
        }
        WMMA_ACC(acc[0][0], af0.v, bfr[0].v, false);
        WMMA_ACC(acc[0][1], af0.v, bfr[1].v, true);
        WMMA_ACC(acc[0][2], af0.v, bfr[2].v, true);
        WMMA_ACC(acc[0][3], af0.v, bfr[3].v, true);
        WMMA_ACC(acc[1][0], af1.v, bfr[0].v, false);
        WMMA_ACC(acc[1][1], af1.v, bfr[1].v, true);
        WMMA_ACC(acc[1][2], af1.v, bfr[2].v, true);
        WMMA_ACC(acc[1][3], af1.v, bfr[3].v, true);
        __syncthreads();
    }

    // epilogue: C layout — VGPR j: lanes 0-15 -> M=j, lanes 16-31 -> M=j+8; N = lane%16
    int rhalf = (lane < 16) ? 0 : 8;
    int ncol = lane & 15;
#pragma unroll
    for (int mi = 0; mi < 2; ++mi) {
#pragma unroll
        for (int c = 0; c < 4; ++c) {
            int col = oBase + wc * 64 + c * 16 + ncol;
            float bv = bias ? bias[col] : 0.0f;
#pragma unroll
            for (int j = 0; j < 8; ++j) {
                int row = mBase + wr * 32 + mi * 16 + j + rhalf;
                if (row < M) {
                    float v = acc[mi][c][j] + bv;
                    if (relu) v = fmaxf(v, 0.0f);
                    C[(size_t)row * EDIM + col] = v;
                }
            }
        }
    }
}

// ---------------- anchor-query attention: 16 queries x 1 head per block ----------------
__global__ __launch_bounds__(256) void attn_a_kernel(
    const float* __restrict__ Q, const float* __restrict__ K, const float* __restrict__ V,
    float* __restrict__ out, int nKeys) {
    __shared__ float Qs[16][HD];
    __shared__ float Ks[64][HD];
    __shared__ float Vs[64][HD];
    __shared__ float ps[16][64];
    __shared__ float Lm[16], Ll[16], La[16];
    int t = threadIdx.x;
    int a0 = blockIdx.x * 16;
    int h = blockIdx.y;
    {
        int r = t >> 4, c = (t & 15) * 4;
        *(float4*)&Qs[r][c] = *(const float4*)&Q[(size_t)(a0 + r) * EDIM + h * HD + c];
    }
    if (t < 16) { Lm[t] = -3.0e38f; Ll[t] = 0.0f; }
    float acc[4] = {0.f, 0.f, 0.f, 0.f};
    __syncthreads();

    for (int c0 = 0; c0 < nKeys; c0 += 64) {
#pragma unroll
        for (int i = 0; i < 4; ++i) {
            int v4 = t + i * 256;
            int r = v4 >> 4, cc = (v4 & 15) * 4;
            int key = c0 + r;
            float4 kv = {0.f, 0.f, 0.f, 0.f}, vv = {0.f, 0.f, 0.f, 0.f};
            if (key < nKeys) {
                kv = *(const float4*)&K[(size_t)key * EDIM + h * HD + cc];
                vv = *(const float4*)&V[(size_t)key * EDIM + h * HD + cc];
            }
            int key2 = key + 64;
            if (key2 < nKeys) {   // prefetch next chunk (global_prefetch_b8)
                __builtin_prefetch(&K[(size_t)key2 * EDIM + h * HD + cc], 0, 1);
                __builtin_prefetch(&V[(size_t)key2 * EDIM + h * HD + cc], 0, 1);
            }
            *(float4*)&Ks[r][cc] = kv;
            *(float4*)&Vs[r][cc] = vv;
        }
        __syncthreads();
#pragma unroll
        for (int i = 0; i < 4; ++i) {
            int idx = t + i * 256;
            int q = idx >> 6, k = idx & 63;
            float s = 0.0f;
#pragma unroll
            for (int d = 0; d < HD; ++d) s += Qs[q][d] * Ks[k][d];
            ps[q][k] = (c0 + k < nKeys) ? s * 0.125f : -3.0e38f;
        }
        __syncthreads();
        if (t < 16) {
            float m_old = Lm[t];
            float cm = -3.0e38f;
            for (int k = 0; k < 64; ++k) cm = fmaxf(cm, ps[t][k]);
            float m_new = fmaxf(m_old, cm);
            float alpha = __expf(m_old - m_new);
            float l = Ll[t] * alpha;
            for (int k = 0; k < 64; ++k) {
                float e = __expf(ps[t][k] - m_new);
                ps[t][k] = e;
                l += e;
            }
            Lm[t] = m_new; Ll[t] = l; La[t] = alpha;
        }
        __syncthreads();
#pragma unroll
        for (int i = 0; i < 4; ++i) {
            int idx = t + i * 256;
            int q = idx >> 6, d = idx & 63;
            float s = 0.0f;
            for (int k = 0; k < 64; ++k) s += ps[q][k] * Vs[k][d];
            acc[i] = acc[i] * La[q] + s;
        }
        __syncthreads();
    }
#pragma unroll
    for (int i = 0; i < 4; ++i) {
        int idx = t + i * 256;
        int q = idx >> 6, d = idx & 63;
        out[(size_t)(a0 + q) * EDIM + h * HD + d] = acc[i] / Ll[q];
    }
}

// ---------------- node-query attention: 256 anchor keys, 1 node/thread ----------------
__global__ __launch_bounds__(256) void attn_e_kernel(
    const float* __restrict__ Q, const float* __restrict__ Kc, const float* __restrict__ Vc,
    float* __restrict__ out, int nQ) {
    __shared__ float Ks[64][HD];
    __shared__ float Vs[64][HD];
    int t = threadIdx.x;
    int n = blockIdx.x * 256 + t;
    int h = blockIdx.y;
    float qv[HD], acc[HD];
#pragma unroll
    for (int d = 0; d < HD; ++d) acc[d] = 0.0f;
    if (n < nQ) {
#pragma unroll
        for (int d = 0; d < HD; ++d) qv[d] = Q[(size_t)n * EDIM + h * HD + d];
    } else {
#pragma unroll
        for (int d = 0; d < HD; ++d) qv[d] = 0.0f;
    }
    float m = -3.0e38f, l = 0.0f;
    for (int c0 = 0; c0 < ANCH; c0 += 64) {
#pragma unroll
        for (int i = 0; i < 4; ++i) {
            int v4 = t + i * 256;
            int r = v4 >> 4, cc = (v4 & 15) * 4;
            *(float4*)&Ks[r][cc] = *(const float4*)&Kc[(size_t)(c0 + r) * EDIM + h * HD + cc];
            *(float4*)&Vs[r][cc] = *(const float4*)&Vc[(size_t)(c0 + r) * EDIM + h * HD + cc];
        }
        __syncthreads();
        if (n < nQ) {
            for (int k = 0; k < 64; ++k) {
                float s = 0.0f;
#pragma unroll
                for (int d = 0; d < HD; ++d) s += qv[d] * Ks[k][d];
                s *= 0.125f;
                float m_new = fmaxf(m, s);
                float corr = __expf(m - m_new);
                float p = __expf(s - m_new);
                l = l * corr + p;
#pragma unroll
                for (int d = 0; d < HD; ++d) acc[d] = acc[d] * corr + p * Vs[k][d];
                m = m_new;
            }
        }
        __syncthreads();
    }
    if (n < nQ) {
        float rl = 1.0f / l;
#pragma unroll
        for (int d = 0; d < HD; ++d) out[(size_t)n * EDIM + h * HD + d] = acc[d] * rl;
    }
}

// =========================================================================
extern "C" void kernel_launch(void* const* d_in, const int* in_sizes, int n_in,
                              void* d_out, int out_size, void* d_ws, size_t ws_size,
                              hipStream_t stream) {
    const float* embeds = (const float*)d_in[0];
    const float* vals   = (const float*)d_in[1];
    const int*   rowsP  = (const int*)d_in[2];
    const int*   colsP  = (const int*)d_in[3];
    const int*   anchor = (const int*)d_in[4];
    const float* Wq = (const float*)d_in[5];
    const float* Wk = (const float*)d_in[6];
    const float* Wv = (const float*)d_in[7];
    const float* Wo = (const float*)d_in[8];
    const float* ln1g = (const float*)d_in[9];
    const float* ln1b = (const float*)d_in[10];
    const float* ln2g = (const float*)d_in[11];
    const float* ln2b = (const float*)d_in[12];
    const float* W1 = (const float*)d_in[13];
    const float* b1 = (const float*)d_in[14];
    const float* W2 = (const float*)d_in[15];
    const float* b2 = (const float*)d_in[16];

    const int Nn  = in_sizes[0] / EDIM;   // 50000
    const int nnz = in_sizes[1];          // 1.6M
    const size_t NE = (size_t)Nn * EDIM;
    const size_t EE = (size_t)EDIM * EDIM;
    const size_t AE = (size_t)ANCH * EDIM;
    const size_t NpadRows = ((size_t)Nn + 127) & ~(size_t)127;  // async loads: no OOB guard
    const size_t NEpad = NpadRows * EDIM;

    // -------- workspace layout --------
    char* base = (char*)d_ws;
    size_t off = 0;
    auto nextbuf = [&](size_t bytes) -> void* {
        void* p = (void*)(base + off);
        off = (off + bytes + 255) & ~(size_t)255;
        return p;
    };
    float* B0 = (float*)nextbuf(NE * 4);
    float* B1 = (float*)nextbuf(NE * 4);
    float* B2 = (float*)nextbuf(NE * 4);
    float* B3 = (float*)nextbuf(NE * 4);
    float* B4 = (float*)nextbuf(NE * 4);
    bf16*  XBF = (bf16*)nextbuf(NEpad * 2);
    bf16*  WBF = (bf16*)nextbuf(12 * EE * 2);
    float* S0 = (float*)nextbuf(AE * 4);  // anchors
    float* S1 = (float*)nextbuf(AE * 4);  // qa
    float* S2 = (float*)nextbuf(AE * 4);  // attn_a
    float* S3 = (float*)nextbuf(AE * 4);  // a2 proj
    float* S4 = (float*)nextbuf(AE * 4);  // anchors2
    float* S5 = (float*)nextbuf(AE * 4);  // ka
    float* S6 = (float*)nextbuf(AE * 4);  // va
    bf16*  SBF = (bf16*)nextbuf(AE * 2);

    auto wslot = [&](int l, int s) { return WBF + ((size_t)(l * 6 + s)) * EE; };
    auto conv = [&](const float* in, bf16* o, size_t n) {
        conv_bf16_kernel<<<(unsigned)((n / 4 + 255) / 256), 256, 0, stream>>>(in, o, (int)n);
    };
    auto gemm = [&](const bf16* Ain, const bf16* Win, float* Cout, int M,
                    const float* bias, int relu) {
        dim3 g((unsigned)((M + 127) / 128), EDIM / 128);
        gemm_bf16_kernel<<<g, 256, 0, stream>>>(Ain, Win, Cout, M, bias, relu);
    };

    // zero the padded tail rows of XBF once per launch (read by async A-tile loads)
    if (NEpad > NE)
        (void)hipMemsetAsync(XBF + NE, 0, (NEpad - NE) * 2, stream);

    // -------- weights -> bf16 (once) --------
    for (int l = 0; l < LGT; ++l) {
        conv(Wq + (size_t)l * EE, wslot(l, 0), EE);
        conv(Wk + (size_t)l * EE, wslot(l, 1), EE);
        conv(Wv + (size_t)l * EE, wslot(l, 2), EE);
        conv(Wo + (size_t)l * EE, wslot(l, 3), EE);
        conv(W1 + (size_t)l * EE, wslot(l, 4), EE);
        conv(W2 + (size_t)l * EE, wslot(l, 5), EE);
    }

    // -------- TopoEncoder: LN + 2 SpMM hops, summed --------
    ln_kernel<<<Nn, 256, 0, stream>>>(embeds, nullptr, nullptr, nullptr, B0, 1.0f);
    (void)hipMemsetAsync(B2, 0, NE * 4, stream);
    unsigned spmmBlocks = (unsigned)(((long long)nnz * 64 + 255) / 256);
    (void)hipMemsetAsync(B1, 0, NE * 4, stream);
    spmm_kernel<<<spmmBlocks, 256, 0, stream>>>(B0, vals, rowsP, colsP, B1, nnz);
    add2_kernel<<<(unsigned)((NE + 255) / 256), 256, 0, stream>>>(B2, B2, B1, (int)NE);
    (void)hipMemsetAsync(B0, 0, NE * 4, stream);
    spmm_kernel<<<spmmBlocks, 256, 0, stream>>>(B1, vals, rowsP, colsP, B0, nnz);
    add2_kernel<<<(unsigned)((NE + 255) / 256), 256, 0, stream>>>(B2, B2, B0, (int)NE);
    // x = B2

    // -------- GraphTransformer layers --------
    for (int l = 0; l < LGT; ++l) {
        float* x = B2;
        conv(x, XBF, NE);
        gemm(XBF, wslot(l, 1), B0, Nn, nullptr, 0);   // Kf = x @ Wk^T
        gemm(XBF, wslot(l, 2), B1, Nn, nullptr, 0);   // Vf = x @ Wv^T
        gemm(XBF, wslot(l, 0), B3, Nn, nullptr, 0);   // Qx = x @ Wq^T

        gather_kernel<<<(unsigned)(AE / 256), 256, 0, stream>>>(x, anchor + l * ANCH, S0);
        conv(S0, SBF, AE);
        gemm(SBF, wslot(l, 0), S1, ANCH, nullptr, 0); // qa = anchors @ Wq^T

        attn_a_kernel<<<dim3(ANCH / 16, HEADS), 256, 0, stream>>>(S1, B0, B1, S2, Nn);
        conv(S2, SBF, AE);
        gemm(SBF, wslot(l, 3), S3, ANCH, nullptr, 0); // a2 = attn @ Wo^T
        add2_kernel<<<(unsigned)(AE / 256), 256, 0, stream>>>(S4, S3, S0, (int)AE);

        conv(S4, SBF, AE);
        gemm(SBF, wslot(l, 1), S5, ANCH, nullptr, 0); // ka
        gemm(SBF, wslot(l, 2), S6, ANCH, nullptr, 0); // va

        attn_e_kernel<<<dim3((unsigned)((Nn + 255) / 256), HEADS), 256, 0, stream>>>(
            B3, S5, S6, B4, Nn);
        conv(B4, XBF, NE);
        gemm(XBF, wslot(l, 3), B0, Nn, nullptr, 0);   // e2 = attn @ Wo^T

        ln_kernel<<<Nn, 256, 0, stream>>>(B0, x, ln1g + l * EDIM, ln1b + l * EDIM, B1, 1.0f);

        conv(B1, XBF, NE);
        gemm(XBF, wslot(l, 4), B3, Nn, b1 + (size_t)l * EDIM, 1);  // h1 = relu(x@W1^T+b1)
        conv(B3, XBF, NE);
        gemm(XBF, wslot(l, 5), B4, Nn, b2 + (size_t)l * EDIM, 1);  // h2 = relu(h1@W2^T+b2)

        ln_kernel<<<Nn, 256, 0, stream>>>(B4, B1, ln2g + l * EDIM, ln2b + l * EDIM, B2,
                                          1.0f /* /SCALE, SCALE=1 */);
    }

    (void)hipMemcpyAsync(d_out, B2, NE * 4, hipMemcpyDeviceToDevice, stream);
}
